// IndividualDecoder_19559281066669
// MI455X (gfx1250) — matrix operand order repaired and checked by reading, not compiled, not run on hardware
//
#include <hip/hip_runtime.h>
#include <hip/hip_bf16.h>

// Problem sizes (match reference)
#define BATCH 8192
#define INF   1024      // IN (K)
#define OUTF  1024      // OUT
#define HIDF  4
#define NTOT  (OUTF*HIDF)   // 4096 = N of fused layer-1 GEMM

// GEMM tiling
#define BM 128
#define BN 128          // 32 MLPs per block
#define BK 32
#define KTILES (INF/BK) // 32

// LDS row stride (bf16 elems): 40 elems = 80 B -> conflict-free b128 frag loads
#define AST 40

typedef __attribute__((ext_vector_type(16))) __bf16 v16bf;
typedef __attribute__((ext_vector_type(8)))  __bf16 v8bf;
typedef __attribute__((ext_vector_type(4)))  __bf16 v4bf;
typedef __attribute__((ext_vector_type(8)))  float  v8f;
typedef __attribute__((ext_vector_type(4)))  float  f32x4;

// Types required by __builtin_amdgcn_global_load_async_to_lds_b128:
//   arg0: addrspace(1) pointer to int vector_size(16), non-const
//   arg1: addrspace(3) pointer (LDS destination)
typedef int v4i __attribute__((vector_size(16)));
typedef __attribute__((address_space(1))) v4i* gvec;
typedef __attribute__((address_space(3))) v4i* lvec;

#if __has_builtin(__builtin_amdgcn_global_load_async_to_lds_b128) && \
    __has_builtin(__builtin_amdgcn_s_wait_asynccnt)
#define USE_ASYNC 1
#define ASYNC_CP16(g, l) \
  __builtin_amdgcn_global_load_async_to_lds_b128((gvec)(void*)(g), (lvec)(void*)(l), 0, 0)
#else
#define USE_ASYNC 0
#endif

// Lane-xor exchange via ds_swizzle (group-of-32 mode: and=0x1F, or=0, xor=mask)
#if __has_builtin(__builtin_amdgcn_ds_swizzle)
template <int PAT>
__device__ __forceinline__ float swz_xor(float v) {
  return __int_as_float(__builtin_amdgcn_ds_swizzle(__float_as_int(v), PAT));
}
#define SWZ_X1(v) swz_xor<0x041F>(v)   // xor_mask = 1
#define SWZ_X2(v) swz_xor<0x081F>(v)   // xor_mask = 2
#else
#define SWZ_X1(v) __shfl_xor((v), 1, 32)
#define SWZ_X2(v) __shfl_xor((v), 2, 32)
#endif

__device__ __forceinline__ v16bf cat8(v8bf lo, v8bf hi) {
  v16bf r;
#pragma unroll
  for (int i = 0; i < 8; ++i) { r[i] = lo[i]; r[i + 8] = hi[i]; }
  return r;
}

// ---------------- conversion kernels (one-shot, bandwidth-bound) -------------

// x (f32 row-major) -> xb (bf16 row-major)
__global__ __launch_bounds__(256) void convert_x(const float* __restrict__ x,
                                                 __bf16* __restrict__ xb) {
  size_t i = (size_t)blockIdx.x * blockDim.x + threadIdx.x;  // one float4 each
  f32x4 v = *(const f32x4*)(x + i * 4);
  v4bf t;
  t[0] = (__bf16)v[0]; t[1] = (__bf16)v[1];
  t[2] = (__bf16)v[2]; t[3] = (__bf16)v[3];
  *(v4bf*)(xb + i * 4) = t;
}

// W1 (OUT,IN,4) f32 -> wb[n][k] bf16 with n = o*4+h  (B^T of the fused GEMM)
__global__ __launch_bounds__(256) void convert_w1(const float* __restrict__ W1,
                                                  __bf16* __restrict__ wb) {
  int idx = blockIdx.x * 256 + threadIdx.x;   // 0 .. 4096*128-1
  int n   = idx >> 7;                         // 0..4095
  int kc  = (idx & 127) << 3;                 // k chunk of 8
  int o = n >> 2, h = n & 3;
  const float* src = W1 + (size_t)o * (INF * HIDF) + (size_t)kc * HIDF + h;
  v8bf t;
#pragma unroll
  for (int j = 0; j < 8; ++j) t[j] = (__bf16)src[j * 4];
  *(v8bf*)(wb + (size_t)n * INF + kc) = t;
}

// ---------------- main GEMM + fused epilogue ---------------------------------

__global__ __launch_bounds__(256) void gemm_mlp(
    const __bf16* __restrict__ xb, const __bf16* __restrict__ wb,
    const float* __restrict__ b1,  const float* __restrict__ W2,
    const float* __restrict__ b2,  float* __restrict__ out)
{
  __shared__ __bf16 As[2][BM * AST];
  __shared__ __bf16 Bs[2][BN * AST];

  const int tid  = threadIdx.x;
  const int lane = tid & 31;
  const int wave = tid >> 5;
  const int wm   = wave & 3;    // 4 waves along M (32 rows each)
  const int wn   = wave >> 2;   // 2 waves along N (64 cols each)
  const int m0   = blockIdx.y * BM;
  const int n0   = blockIdx.x * BN;
  const int l15  = lane & 15;
  const int lhi  = lane >> 4;

  // Staging: A and B tiles are both 128 rows x 64B strips of bf16.
  // 512 16-byte chunks each; every thread moves 2 A-chunks + 2 B-chunks.
  auto stage = [&](int buf, int kt) {
    const int k0 = kt * BK;
#pragma unroll
    for (int half = 0; half < 2; ++half) {
      const int t   = tid + half * 256;
      const int row = t >> 2;
      const int ch  = (t & 3) << 3;  // bf16 elem offset of 16B chunk
      const __bf16* ga = xb + (size_t)(m0 + row) * INF + k0 + ch;
      const __bf16* gb = wb + (size_t)(n0 + row) * INF + k0 + ch;
      __bf16* la = &As[buf][row * AST + ch];
      __bf16* lb = &Bs[buf][row * AST + ch];
#if USE_ASYNC
      ASYNC_CP16(ga, la);
      ASYNC_CP16(gb, lb);
#else
      *(v8bf*)la = *(const v8bf*)ga;
      *(v8bf*)lb = *(const v8bf*)gb;
#endif
    }
  };

  // A frag per ISA 16-bit A 16x32 layout: lanes<16 -> K{0..7,16..23}; lanes>=16 -> K{8..15,24..31}
  auto loadA = [&](int buf, int mi) -> v16bf {
    const __bf16* p = &As[buf][(wm * 32 + mi * 16 + l15) * AST + lhi * 8];
    return cat8(*(const v8bf*)p, *(const v8bf*)(p + 16));
  };
  // B frag: lane(&15) = column; lanes<16 hold K{0..7,16..23}, lanes>=16 hold K{8..15,24..31}
  auto loadB = [&](int buf, int ni) -> v16bf {
    const __bf16* p = &Bs[buf][(wn * 64 + ni * 16 + l15) * AST + lhi * 8];
    return cat8(*(const v8bf*)p, *(const v8bf*)(p + 16));
  };

  v8f acc[2][4];
#pragma unroll
  for (int i = 0; i < 2; ++i)
#pragma unroll
    for (int j = 0; j < 4; ++j)
#pragma unroll
      for (int e = 0; e < 8; ++e) acc[i][j][e] = 0.0f;

  stage(0, 0);
#if USE_ASYNC
  __builtin_amdgcn_s_wait_asynccnt(0);
#endif
  __syncthreads();

  for (int kt = 0; kt < KTILES; ++kt) {
    const int buf = kt & 1;
    if (kt + 1 < KTILES) stage(buf ^ 1, kt + 1);

    v16bf a[2], b[4];
#pragma unroll
    for (int mi = 0; mi < 2; ++mi) a[mi] = loadA(buf, mi);
#pragma unroll
    for (int ni = 0; ni < 4; ++ni) b[ni] = loadB(buf, ni);

#pragma unroll
    for (int mi = 0; mi < 2; ++mi)
#pragma unroll
      for (int ni = 0; ni < 4; ++ni)
        acc[mi][ni] = __builtin_amdgcn_wmma_f32_16x16x32_bf16(
            false, a[mi], false, b[ni], (short)0, acc[mi][ni], false, false);

#if USE_ASYNC
    __builtin_amdgcn_s_wait_asynccnt(0);
#endif
    __syncthreads();
  }

  // Epilogue: h = acc + b1; LeakyReLU(0.1); out[b,o] = sum_h h*W2[o,h] + b2[o]
  // C layout: VGPR r -> rows (r, r+8); lane(&15) -> column within 16-tile.
#pragma unroll
  for (int ni = 0; ni < 4; ++ni) {
    const int nIdx = n0 + wn * 64 + ni * 16 + l15;  // flat column = o*4+h
    const int oIdx = nIdx >> 2;
    const float b1v = b1[nIdx];
    const float w2v = W2[nIdx];   // W2 is (OUT,4) flat, indexed by nIdx
    const float b2v = b2[oIdx];
#pragma unroll
    for (int mi = 0; mi < 2; ++mi) {
      const int rowBase = m0 + wm * 32 + mi * 16 + lhi * 8;
#pragma unroll
      for (int r = 0; r < 8; ++r) {
        float v = acc[mi][ni][r] + b1v;
        v = (v >= 0.0f) ? v : 0.1f * v;     // LeakyReLU(0.1)
        v *= w2v;
        v += SWZ_X1(v);                     // reduce the 4 h-lanes
        v += SWZ_X2(v);
        if ((lane & 3) == 0)
          out[(size_t)(rowBase + r) * OUTF + oIdx] = v + b2v;
      }
    }
  }
}

// ---------------- launcher ---------------------------------------------------

extern "C" void kernel_launch(void* const* d_in, const int* in_sizes, int n_in,
                              void* d_out, int out_size, void* d_ws, size_t ws_size,
                              hipStream_t stream) {
  const float* x  = (const float*)d_in[0];
  const float* W1 = (const float*)d_in[1];
  const float* b1 = (const float*)d_in[2];
  const float* W2 = (const float*)d_in[3];
  const float* b2 = (const float*)d_in[4];
  float* out = (float*)d_out;

  __bf16* xb = (__bf16*)d_ws;                       // 8192*1024 bf16 = 16 MB
  __bf16* wb = xb + (size_t)BATCH * INF;            // 4096*1024 bf16 =  8 MB

  // 1) one-shot precision conversion + B transpose into workspace
  convert_x <<<dim3((BATCH * INF / 4) / 256), dim3(256), 0, stream>>>(x, xb);
  convert_w1<<<dim3((NTOT * (INF / 8)) / 256), dim3(256), 0, stream>>>(W1, wb);

  // 2) fused GEMM + LeakyReLU + 4-wide contraction
  dim3 grid(NTOT / BN, BATCH / BM);   // (32, 64)
  gemm_mlp<<<grid, dim3(256), 0, stream>>>(xb, wb, b1, W2, b2, out);
}
